// VanillaRNN_58385785422057
// MI455X (gfx1250) — matrix operand order repaired and checked by reading, not compiled, not run on hardware
//
#include <hip/hip_runtime.h>

// ---------------------------------------------------------------------------
// Problem constants (VanillaRNN reference)
// ---------------------------------------------------------------------------
constexpr int V  = 1024;
constexpr int E  = 256;
constexpr int H  = 512;
constexpr int Bs = 64;
constexpr int S  = 2048;
constexpr int R  = Bs * S;          // 131072 time-major rows (r = t*Bs + b)

// ---------------------------------------------------------------------------
// Vector types for WMMA / wide loads
// ---------------------------------------------------------------------------
typedef __bf16        bf16_t;
typedef bf16_t        v16bf __attribute__((ext_vector_type(16)));
typedef float         v8f   __attribute__((ext_vector_type(8)));
typedef unsigned int  u32x4 __attribute__((ext_vector_type(4)));
typedef int           i32x4 __attribute__((ext_vector_type(4)));
typedef int           i32x8 __attribute__((ext_vector_type(8)));

union FragB16 { v16bf v; u32x4 q[2]; unsigned short s[16]; };

__device__ inline unsigned short f32_to_bf16_bits(float f) {
  union { float f; unsigned u; } c; c.f = f;
  unsigned r = c.u + 0x7FFFu + ((c.u >> 16) & 1u);   // round-to-nearest-even
  return (unsigned short)(r >> 16);
}
__device__ inline float bf16_bits_to_f32(unsigned short b) {
  union { unsigned u; float f; } c; c.u = ((unsigned)b) << 16;
  return c.f;
}

// A operand, 16x32 bf16 (ISA 7.12.2): lane m = lane&15, half = lane>>4,
// element j<8  -> k = k0 + 8*half + j          (16B contiguous)
// element j>=8 -> k = k0 + 16 + 8*half + (j-8) (16B contiguous)
__device__ inline v16bf loadA(const unsigned short* rowptr, int k0, int half) {
  FragB16 r;
  const unsigned short* p = rowptr + k0 + half * 8;
  r.q[0] = *(const u32x4*)(p);
  r.q[1] = *(const u32x4*)(p + 16);
  return r.v;
}

// B operand, 32x16 bf16: lane n = lane&15, same k order as A:
// element j<8 -> k = k0 + 8*half + j, j>=8 -> k = k0 + 16 + 8*half + (j-8)
// (W kept row-major [N x K] so this is W[n, k...])
__device__ inline v16bf loadB(const unsigned short* colptr, int k0, int half) {
  FragB16 r;
  const unsigned short* p = colptr + k0 + half * 8;
  r.q[0] = *(const u32x4*)(p);
  r.q[1] = *(const u32x4*)(p + 16);
  return r.v;
}

__device__ inline v8f wmma_bf16(v16bf a, v16bf b, v8f c) {
  return __builtin_amdgcn_wmma_f32_16x16x32_bf16(
      /*neg_a=*/false, a, /*neg_b=*/false, b,
      /*c_mod=*/(short)0, c, /*reuse_a=*/false, /*reuse_b=*/false);
}

// ---------------------------------------------------------------------------
// TDM: async load of one timestep's P tile (64 rows x 512 cols, bf16 = 64KB)
// from global into LDS. D# per CDNA5 ISA.
// clang-23 lane: 6-arg builtin (u32x4, i32x8, i32x4, i32x4, i32x8, i32 cpol).
// ---------------------------------------------------------------------------
__device__ inline void tdm_load_p_tile(const unsigned short* Pbase, int t,
                                       unsigned short* ldsdst) {
  unsigned long long ga = (unsigned long long)(uintptr_t)Pbase
                        + (unsigned long long)t * (unsigned long long)(Bs * H * 2);
  unsigned lds = (unsigned)(uintptr_t)ldsdst;

  u32x4 g0;
  g0[0] = 1u;                                            // count=1 (valid), user mode
  g0[1] = lds;                                           // lds_addr
  g0[2] = (unsigned)(ga & 0xFFFFFFFFu);                  // global_addr[31:0]
  g0[3] = (unsigned)((ga >> 32) & 0x01FFFFFFu)           // global_addr[56:32]
        | (2u << 30);                                    // type = 2 ("image")

  i32x8 g1;
  g1[0] = (1 << 16);                                     // data_size=1 (2 bytes)
  g1[1] = (int)((H & 0xFFFF) << 16);                     // tensor_dim0[15:0] @ [31:16]
  g1[2] = (int)(((R & 0xFFFF) << 16) | ((H >> 16) & 0xFFFF)); // dim1 lo | dim0 hi
  g1[3] = (int)(((H & 0xFFFF) << 16) | ((R >> 16) & 0xFFFF)); // tile_dim0=H | dim1 hi
  g1[4] = Bs;                                            // tile_dim1=64, tile_dim2=0
  g1[5] = H;                                             // tensor_dim0_stride[31:0]
  g1[6] = (int)((H & 0xFFFF) << 16);                     // dim1_stride[15:0] @ [31:16]
  g1[7] = 0;                                             // dim1_stride hi
  i32x4 z4 = {0, 0, 0, 0};                               // 2D tensor: groups 2/3 zero
  i32x8 z8 = {0, 0, 0, 0, 0, 0, 0, 0};
  __builtin_amdgcn_tensor_load_to_lds(g0, g1, z4, z4, z8, 0);
}

// ---------------------------------------------------------------------------
// Kernel 1: f32 -> bf16 weight conversion (grid-stride)
// ---------------------------------------------------------------------------
__global__ void k_cvt(const float* __restrict__ src, unsigned short* __restrict__ dst,
                      int n) {
  for (int i = blockIdx.x * blockDim.x + threadIdx.x; i < n;
       i += gridDim.x * blockDim.x)
    dst[i] = f32_to_bf16_bits(src[i]);
}

// Kernel 2: fused biases (bias0 = bxh0+bhh0, bias1 = bxh1+bhh1)
__global__ void k_bias(const float* bx0, const float* bh0,
                       const float* bx1, const float* bh1,
                       float* b0, float* b1) {
  int i = threadIdx.x;           // blockDim = H
  b0[i] = bx0[i] + bh0[i];
  b1[i] = bx1[i] + bh1[i];
}

// ---------------------------------------------------------------------------
// Kernel 3: P0 = embed-gather(x) @ Wxh0^T + bias0   (bf16 out, time-major rows)
// One wave computes a 64x16 strip (4 M-tiles reuse one B fragment).
// ---------------------------------------------------------------------------
__global__ void k_gemm_p0(const int* __restrict__ x,
                          const unsigned short* __restrict__ embb,
                          const unsigned short* __restrict__ Wb,   // [H x E]
                          const float* __restrict__ bias,
                          unsigned short* __restrict__ P) {
  const int lane  = threadIdx.x & 31;
  const int wave  = threadIdx.x >> 5;
  const int group = blockIdx.x * (blockDim.x >> 5) + wave;
  const int NT = H / 16;                 // 32
  const int nt = group % NT;
  const int r0 = (group / NT) * 64;      // 64 rows == one timestep (all b)
  const int t  = r0 >> 6;
  const int half = lane >> 4;
  const int col  = nt * 16 + (lane & 15);
  const unsigned short* colptr = Wb + col * E;

  const unsigned short* rowp[4];
#pragma unroll
  for (int i = 0; i < 4; ++i) {
    int b   = i * 16 + (lane & 15);
    int tok = x[b * S + t];              // x is [B, S]
    rowp[i] = embb + tok * E;
  }

  v8f acc[4] = {};
#pragma unroll
  for (int k0 = 0; k0 < E; k0 += 32) {
    __builtin_prefetch(colptr + k0 + 64, 0, 0);
    v16bf bf = loadB(colptr, k0, half);
#pragma unroll
    for (int i = 0; i < 4; ++i)
      acc[i] = wmma_bf16(loadA(rowp[i], k0, half), bf, acc[i]);
  }

  float bv = bias[col];
#pragma unroll
  for (int i = 0; i < 4; ++i)
#pragma unroll
    for (int rr = 0; rr < 8; ++rr) {
      int row = r0 + i * 16 + rr + 8 * half;
      P[(size_t)row * H + col] = f32_to_bf16_bits(acc[i][rr] + bv);
    }
}

// ---------------------------------------------------------------------------
// Kernel 4: P = Ain @ W^T + bias  (bf16 A [R x H], W [H x H]) — layer-1 input proj
// ---------------------------------------------------------------------------
__global__ void k_gemm_proj(const unsigned short* __restrict__ Ain,
                            const unsigned short* __restrict__ Wb,
                            const float* __restrict__ bias,
                            unsigned short* __restrict__ P) {
  const int lane  = threadIdx.x & 31;
  const int wave  = threadIdx.x >> 5;
  const int group = blockIdx.x * (blockDim.x >> 5) + wave;
  const int NT = H / 16;
  const int nt = group % NT;
  const int r0 = (group / NT) * 64;
  const int half = lane >> 4;
  const int col  = nt * 16 + (lane & 15);
  const unsigned short* colptr = Wb + col * H;

  const unsigned short* rowp[4];
#pragma unroll
  for (int i = 0; i < 4; ++i)
    rowp[i] = Ain + (size_t)(r0 + i * 16 + (lane & 15)) * H;

  v8f acc[4] = {};
#pragma unroll 4
  for (int k0 = 0; k0 < H; k0 += 32) {
    __builtin_prefetch(colptr + k0 + 64, 0, 0);
    v16bf bf = loadB(colptr, k0, half);
#pragma unroll
    for (int i = 0; i < 4; ++i)
      acc[i] = wmma_bf16(loadA(rowp[i], k0, half), bf, acc[i]);
  }

  float bv = bias[col];
#pragma unroll
  for (int i = 0; i < 4; ++i)
#pragma unroll
    for (int rr = 0; rr < 8; ++rr) {
      int row = r0 + i * 16 + rr + 8 * half;
      P[(size_t)row * H + col] = f32_to_bf16_bits(acc[i][rr] + bv);
    }
}

// ---------------------------------------------------------------------------
// Kernel 5: recurrent scan for one layer. Single 1024-thread workgroup
// (32 waves, one WGP). h state double-buffered in LDS; P tiles TDM-prefetched
// one step ahead into LDS; Whh stays L2-resident.
//   h_new[b,h] = tanh(P[t*64+b, h] + sum_k h_prev[b,k] * Whh[h,k])
// ---------------------------------------------------------------------------
__global__ void __launch_bounds__(1024)
k_scan(const unsigned short* __restrict__ P,     // [R x H] bf16, bias folded in
       const unsigned short* __restrict__ Whh,   // [H x H] bf16
       unsigned short* __restrict__ h_all,       // [R x H] bf16 (for next phase)
       float* __restrict__ h_final) {            // [Bs x H] f32 (d_out slice)
  extern __shared__ unsigned short smem[];
  unsigned short* hbuf0 = smem;                  // 64x512 bf16 = 64KB
  unsigned short* hbuf1 = smem + Bs * H;
  unsigned short* pbuf0 = smem + 2 * Bs * H;
  unsigned short* pbuf1 = smem + 3 * Bs * H;     // total 256KB dynamic LDS

  const int lane = threadIdx.x & 31;
  const int wave = threadIdx.x >> 5;             // 0..31 == nt strip
  const int half = lane >> 4;
  const int col  = wave * 16 + (lane & 15);      // hidden index 0..511
  const unsigned short* colptr = Whh + col * H;

  for (int i = threadIdx.x; i < Bs * H; i += blockDim.x) hbuf0[i] = 0;

  if (wave == 0) {
    tdm_load_p_tile(P, 0, pbuf0);
    __builtin_amdgcn_s_wait_tensorcnt(0);
  }
  __syncthreads();

  for (int t = 0; t < S; ++t) {
    const int cur = t & 1;
    unsigned short* hp = cur ? hbuf1 : hbuf0;
    unsigned short* hn = cur ? hbuf0 : hbuf1;
    unsigned short* pc = cur ? pbuf1 : pbuf0;
    unsigned short* pn = cur ? pbuf0 : pbuf1;

    // kick off next step's P tile while we do the recurrent GEMM
    if (wave == 0 && (t + 1) < S) tdm_load_p_tile(P, t + 1, pn);

    v8f acc[4] = {};
#pragma unroll 4
    for (int k0 = 0; k0 < H; k0 += 32) {
      v16bf bf = loadB(colptr, k0, half);        // global (L2-hot weights)
#pragma unroll
      for (int i = 0; i < 4; ++i) {
        const unsigned short* arow = hp + (i * 16 + (lane & 15)) * H;
        acc[i] = wmma_bf16(loadA(arow, k0, half), bf, acc[i]); // LDS ds_load_b128
      }
    }

#pragma unroll
    for (int i = 0; i < 4; ++i)
#pragma unroll
      for (int rr = 0; rr < 8; ++rr) {
        int b = i * 16 + rr + 8 * half;
        float v = tanhf(acc[i][rr] + bf16_bits_to_f32(pc[b * H + col]));
        unsigned short vb = f32_to_bf16_bits(v);
        hn[b * H + col] = vb;
        h_all[((size_t)t * Bs + b) * H + col] = vb;
        if (t == S - 1) h_final[b * H + col] = v;
      }

    if (wave == 0) __builtin_amdgcn_s_wait_tensorcnt(0); // P[t+1] landed
    __syncthreads();                                     // publish hn + pn
  }
}

// ---------------------------------------------------------------------------
// Kernel 6: logits = h1_all @ Wout^T + bout, scattered to batch-major output.
// Nontemporal stores: 512MB stream must not evict L2-resident operands.
// ---------------------------------------------------------------------------
__global__ void k_gemm_out(const unsigned short* __restrict__ Ain,  // [R x H]
                           const unsigned short* __restrict__ Wb,   // [V x H]
                           const float* __restrict__ bout,
                           float* __restrict__ out) {               // [B, S, V]
  const int lane  = threadIdx.x & 31;
  const int wave  = threadIdx.x >> 5;
  const int group = blockIdx.x * (blockDim.x >> 5) + wave;
  const int NT = V / 16;                 // 64
  const int nt = group % NT;
  const int r0 = (group / NT) * 64;      // one timestep worth of rows
  const int t  = r0 >> 6;
  const int half = lane >> 4;
  const int col  = nt * 16 + (lane & 15);
  const unsigned short* colptr = Wb + col * H;

  const unsigned short* rowp[4];
#pragma unroll
  for (int i = 0; i < 4; ++i)
    rowp[i] = Ain + (size_t)(r0 + i * 16 + (lane & 15)) * H;

  v8f acc[4] = {};
#pragma unroll 4
  for (int k0 = 0; k0 < H; k0 += 32) {
    __builtin_prefetch(colptr + k0 + 64, 0, 0);
    v16bf bf = loadB(colptr, k0, half);
#pragma unroll
    for (int i = 0; i < 4; ++i)
      acc[i] = wmma_bf16(loadA(rowp[i], k0, half), bf, acc[i]);
  }

  float bv = bout[col];
#pragma unroll
  for (int i = 0; i < 4; ++i)
#pragma unroll
    for (int rr = 0; rr < 8; ++rr) {
      int b = i * 16 + rr + 8 * half;                    // row & 63
      float* dst = out + ((size_t)b * S + t) * V + col;  // batch-major
      __builtin_nontemporal_store(acc[i][rr] + bv, dst);
    }
}

// ---------------------------------------------------------------------------
// Host: workspace layout + launch pipeline (all on `stream`, graph-safe)
// ---------------------------------------------------------------------------
extern "C" void kernel_launch(void* const* d_in, const int* in_sizes, int n_in,
                              void* d_out, int out_size, void* d_ws, size_t ws_size,
                              hipStream_t stream) {
  const int*   x    = (const int*)  d_in[0];
  const float* emb  = (const float*)d_in[1];
  const float* Wxh0 = (const float*)d_in[2];
  const float* bxh0 = (const float*)d_in[3];
  const float* Whh0 = (const float*)d_in[4];
  const float* bhh0 = (const float*)d_in[5];
  const float* Wxh1 = (const float*)d_in[6];
  const float* bxh1 = (const float*)d_in[7];
  const float* Whh1 = (const float*)d_in[8];
  const float* bhh1 = (const float*)d_in[9];
  const float* Wout = (const float*)d_in[10];
  const float* bout = (const float*)d_in[11];

  char* ws = (char*)d_ws;
  size_t off = 0;
  auto take = [&](size_t bytes) -> char* {
    char* p = ws + off;
    off += bytes;
    off = (off + 255) & ~(size_t)255;
    return p;
  };
  unsigned short* emb_b  = (unsigned short*)take((size_t)V * E * 2);
  unsigned short* wxh0_b = (unsigned short*)take((size_t)H * E * 2);
  unsigned short* whh0_b = (unsigned short*)take((size_t)H * H * 2);
  unsigned short* wxh1_b = (unsigned short*)take((size_t)H * H * 2);
  unsigned short* whh1_b = (unsigned short*)take((size_t)H * H * 2);
  unsigned short* wout_b = (unsigned short*)take((size_t)V * H * 2);
  float*          bias0  = (float*)take((size_t)H * 4);
  float*          bias1  = (float*)take((size_t)H * 4);
  unsigned short* Pbuf   = (unsigned short*)take((size_t)R * H * 2); // P0 then P1
  unsigned short* h0_all = (unsigned short*)take((size_t)R * H * 2);
  unsigned short* h1_all = (unsigned short*)take((size_t)R * H * 2);

  float* logits = (float*)d_out;                         // [B, S, V]
  float* hout0  = logits + (size_t)Bs * S * V;           // [B, H]
  float* hout1  = hout0 + (size_t)Bs * H;                // [B, H]

  // 1) weights -> bf16, fused biases
  k_cvt<<<512,  256, 0, stream>>>(emb,  emb_b,  V * E);
  k_cvt<<<256,  256, 0, stream>>>(Wxh0, wxh0_b, H * E);
  k_cvt<<<512,  256, 0, stream>>>(Whh0, whh0_b, H * H);
  k_cvt<<<512,  256, 0, stream>>>(Whh1, whh1_b, H * H);
  k_cvt<<<512,  256, 0, stream>>>(Wxh1, wxh1_b, H * H);
  k_cvt<<<1024, 256, 0, stream>>>(Wout, wout_b, V * H);
  k_bias<<<1, H, 0, stream>>>(bxh0, bhh0, bxh1, bhh1, bias0, bias1);

  // 2) hoisted input projection, layer 0 (embedding gather fused in)
  k_gemm_p0<<<8192, 256, 0, stream>>>(x, emb_b, wxh0_b, bias0, Pbuf);

  // 3) serial scan layer 0 (single WGP, LDS state, TDM P-prefetch)
  k_scan<<<1, 1024, 256 * 1024, stream>>>(Pbuf, whh0_b, h0_all, hout0);

  // 4) hoisted input projection, layer 1 (reuses Pbuf)
  k_gemm_proj<<<8192, 256, 0, stream>>>(h0_all, wxh1_b, bias1, Pbuf);

  // 5) serial scan layer 1
  k_scan<<<1, 1024, 256 * 1024, stream>>>(Pbuf, whh1_b, h1_all, hout1);

  // 6) output projection, batch-major nontemporal stores
  k_gemm_out<<<16384, 256, 0, stream>>>(h1_all, wout_b, bout, logits);
}
